// GNNModel_31267361915421
// MI455X (gfx1250) — matrix operand order repaired and checked by reading, not compiled, not run on hardware
//
#include <hip/hip_runtime.h>
#include <hip/hip_bf16.h>

// ---------------------------------------------------------------------------
// GINE GNN forward for MI455X (gfx1250):
//   - bf16 WMMA (v_wmma_f32_16x16x32_bf16) for every K in {64,128} GEMM
//   - Tensor Data Mover (tensor_load_to_lds) stages bf16 A-tiles into LDS
//     with hardware pad (row stride K+8 halves) and OOB zero-fill
// ---------------------------------------------------------------------------

#define NNODES 100000
#define NEDGES 600000
#define NGRAPH 4096
#define HID    128
#define BN_EPS 1e-5f

typedef __attribute__((ext_vector_type(16))) __bf16         v16bf;
typedef __attribute__((ext_vector_type(8)))  float          v8f;
typedef __attribute__((ext_vector_type(8)))  unsigned short v8us;
typedef __attribute__((ext_vector_type(16))) unsigned short v16us;
typedef __attribute__((ext_vector_type(4)))  unsigned int   v4u;
typedef __attribute__((ext_vector_type(8)))  int            v8i;
typedef __attribute__((ext_vector_type(4)))  int            v4i;

#if defined(__has_builtin)
#if __has_builtin(__builtin_amdgcn_tensor_load_to_lds) && \
    __has_builtin(__builtin_amdgcn_s_wait_tensorcnt)
#define USE_TDM 1
#endif
#endif
#ifndef USE_TDM
#define USE_TDM 0
#endif

__device__ __forceinline__ unsigned short f2bf(float f) {
  unsigned int u = __float_as_uint(f);
  u += 0x7fffu + ((u >> 16) & 1u);          // round-to-nearest-even
  return (unsigned short)(u >> 16);
}

// ---------------------------------------------------------------------------
// Generic fused GEMM: Out[M,N] = act(A[M,K] @ W[K,N] + bias)
// K in {64,128}, N multiple of 16 (<=128). 128 threads = 4 waves, BM=64.
// A tile in LDS with row stride (K+8) halves (TDM pad or manual staging);
// W staged transposed at fixed 136-half stride. Fragments per ISA 7.12.2.
// ---------------------------------------------------------------------------
#define KPAD 136

template<int A_IS_BF16, int OUT_BF16, int RELU>
__global__ __launch_bounds__(128)
void wmma_gemm_bias(const void* __restrict__ Ain, const float* __restrict__ W,
                    const float* __restrict__ bias, void* __restrict__ Out,
                    int M, int K, int N)
{
  __shared__ __align__(16) unsigned short sAf[64 * KPAD];   // stride K+8 halves
  __shared__ __align__(16) unsigned short sWt[128][KPAD];   // [n][k]
  __shared__ float sB[128];

  const int tid     = threadIdx.x;
  const int aStride = K + 8;                // halves per A row in LDS
  const int m0      = blockIdx.x * 64;

  // stage weights (transposed) + bias, converting fp32 -> bf16
  for (int i = tid; i < K * N; i += 128) {
    int k = i / N, n = i % N;
    sWt[n][k] = f2bf(W[i]);
  }
  for (int i = tid; i < N; i += 128) sB[i] = bias[i];

#if USE_TDM
  if (A_IS_BF16) {
    // Tensor DMA: 2D tile, row = K bf16, 64 rows, tensor_dim1 = M - m0 so the
    // M-tail reads OOB -> zero-filled. pad: one row interval, 4 DWORDs (8
    // halves) -> LDS row stride = K + 8 halves.
    if (tid < 32) {   // wave 0 issues the descriptor (EXEC ignored by TDM)
      unsigned long long ga =
          (unsigned long long)(uintptr_t)((const unsigned short*)Ain +
                                          (size_t)m0 * (size_t)K);
      unsigned lds = (unsigned)(uintptr_t)&sAf[0];      // low 32b = LDS offset
      unsigned rows = (unsigned)(M - m0);
      unsigned pi = (K == 128) ? 5u : 4u;               // 64 / 32 DWORD interval
      v4u g0 = { 1u,                                    // count=1 (valid D#)
                 lds,
                 (unsigned)ga,
                 (unsigned)((ga >> 32) & 0x01FFFFFFu) | (2u << 30) }; // type=2
      v8i g1 = { (int)((1u << 16) |                     // data_size = 2B
                       (1u << 20) |                     // pad_enable
                       (pi << 22) |                     // pad_interval
                       (3u << 25)),                     // pad_amount = 4 DW
                 (int)((unsigned)K << 16),              // tensor_dim0 = K
                 (int)((rows & 0xFFFFu) << 16),         // tensor_dim1 lo
                 (int)(((rows >> 16) & 0xFFFFu) |       // tensor_dim1 hi
                       ((unsigned)K << 16)),            // tile_dim0 = K
                 64,                                    // tile_dim1 = 64 rows
                 K,                                     // tensor_dim0_stride
                 0, 0 };
      v4i g2 = { 0, 0, 0, 0 };
      v4i g3 = { 0, 0, 0, 0 };
#if __clang_major__ >= 23
      v8i g4 = { 0, 0, 0, 0, 0, 0, 0, 0 };
      __builtin_amdgcn_tensor_load_to_lds(g0, g1, g2, g3, g4, 0);
#else
      __builtin_amdgcn_tensor_load_to_lds(g0, g1, g2, g3, 0);
#endif
      __builtin_amdgcn_s_wait_tensorcnt(0);
    }
  } else
#endif
  {
    if (A_IS_BF16) {
      const unsigned short* A = (const unsigned short*)Ain;
      for (int i = tid; i < 64 * K; i += 128) {
        int r = i / K, c = i % K, gr = m0 + r;
        sAf[r * aStride + c] = (gr < M) ? A[(long)gr * K + c] : (unsigned short)0;
      }
    } else {
      const float* A = (const float*)Ain;
      for (int i = tid; i < 64 * K; i += 128) {
        int r = i / K, c = i % K, gr = m0 + r;
        sAf[r * aStride + c] = (gr < M) ? f2bf(A[(long)gr * K + c]) : (unsigned short)0;
      }
    }
  }
  __syncthreads();

  const int wave   = tid >> 5;
  const int lane   = tid & 31;
  const int laneHi = lane >> 4;      // 0: lanes 0-15, 1: lanes 16-31
  const int l16    = lane & 15;
  const int rowL   = wave * 16 + l16;   // local A row this lane fragments
  const int ntiles = N >> 4;
  const unsigned short* aRow = &sAf[rowL * aStride];

  for (int nt = 0; nt < ntiles; ++nt) {
    const int col = nt * 16 + l16;
    v8f acc = {0.f, 0.f, 0.f, 0.f, 0.f, 0.f, 0.f, 0.f};
    for (int k0 = 0; k0 < K; k0 += 32) {
      // A 16x32 bf16 frag: lanes 0-15 -> K {0..7,16..23}; lanes 16-31 -> +8
      const unsigned short* pa = aRow + k0 + laneHi * 8;
      v8us a_lo = *(const v8us*)pa;
      v8us a_hi = *(const v8us*)(pa + 16);
      v16us av = __builtin_shufflevector(a_lo, a_hi,
          0,1,2,3,4,5,6,7,8,9,10,11,12,13,14,15);
      // B 32x16 bf16 frag: lanes 0-15 -> K {0..7,16..23}; lanes 16-31 -> +8
      const unsigned short* pb = &sWt[col][k0 + laneHi * 8];
      v8us b_lo = *(const v8us*)pb;
      v8us b_hi = *(const v8us*)(pb + 16);
      v16us bv = __builtin_shufflevector(b_lo, b_hi,
          0,1,2,3,4,5,6,7,8,9,10,11,12,13,14,15);
      acc = __builtin_amdgcn_wmma_f32_16x16x32_bf16(
          false, __builtin_bit_cast(v16bf, av),
          false, __builtin_bit_cast(v16bf, bv),
          (short)0, acc, false, false);
    }
    // C/D layout: VGPR v -> row v (lanes 0-15) / v+8 (lanes 16-31)
#pragma unroll
    for (int v = 0; v < 8; ++v) {
      int gr = m0 + wave * 16 + v + laneHi * 8;
      if (gr < M) {
        float val = acc[v] + sB[col];
        if (RELU) val = fmaxf(val, 0.f);
        if (OUT_BF16)
          ((unsigned short*)Out)[(long)gr * N + col] = f2bf(val);
        else
          ((float*)Out)[(long)gr * N + col] = val;
      }
    }
  }
}

// ---------------------------------------------------------------------------
// Edge MLP layer 1: h1[e,j] = relu(sum_{k<3} ea[e,k]*W1[k,j] + b1[j]) -> bf16
// ---------------------------------------------------------------------------
__global__ void edge_layer1(const float* __restrict__ eattr,
                            const float* __restrict__ W1,
                            const float* __restrict__ b1,
                            unsigned short* __restrict__ out)
{
  long long idx = (long long)blockIdx.x * blockDim.x + threadIdx.x;
  if (idx >= (long long)NEDGES * HID) return;
  int e = (int)(idx >> 7), j = (int)(idx & 127);
  float a0 = eattr[e * 3 + 0], a1 = eattr[e * 3 + 1], a2 = eattr[e * 3 + 2];
  float s = b1[j] + a0 * W1[j] + a1 * W1[HID + j] + a2 * W1[2 * HID + j];
  out[idx] = f2bf(fmaxf(s, 0.f));
}

// msg = relu(x[src] + ea_lin); agg[dst] += msg  (float atomics)
__global__ void msg_scatter(const float* __restrict__ x,
                            const float* __restrict__ ealin,
                            const long long* __restrict__ src,
                            const long long* __restrict__ dst,
                            float* __restrict__ agg, int din)
{
  long long idx = (long long)blockIdx.x * blockDim.x + threadIdx.x;
  if (idx >= (long long)NEDGES * din) return;
  int e = (int)(idx / din), c = (int)(idx % din);
  long long s = src[e], d = dst[e];
  float v = fmaxf(x[s * din + c] + ealin[idx], 0.f);
  atomicAdd(&agg[d * din + c], v);
}

// u = (1+eps)*x + agg
__global__ void make_u(const float* __restrict__ x, const float* __restrict__ agg,
                       const float* __restrict__ eps, float* __restrict__ u,
                       long long n)
{
  long long idx = (long long)blockIdx.x * blockDim.x + threadIdx.x;
  if (idx >= n) return;
  u[idx] = (1.f + eps[0]) * x[idx] + agg[idx];
}

// per-feature sum / sum-of-squares over nodes (128 features)
__global__ void bn_stats(const float* __restrict__ y, float* __restrict__ bsum,
                         float* __restrict__ bsq)
{
  int tg = blockIdx.x * blockDim.x + threadIdx.x;
  int col = tg & 127;
  int r0 = tg >> 7;
  int rstride = (gridDim.x * blockDim.x) >> 7;
  float s = 0.f, s2 = 0.f;
  for (int r = r0; r < NNODES; r += rstride) {
    float v = y[(long)r * HID + col];
    s += v; s2 += v * v;
  }
  atomicAdd(&bsum[col], s);
  atomicAdd(&bsq[col], s2);
}

// h = relu((y-mean)*rsqrt(var+eps)*g + b)
__global__ void bn_apply(const float* __restrict__ y, const float* __restrict__ bsum,
                         const float* __restrict__ bsq, const float* __restrict__ g,
                         const float* __restrict__ b, float* __restrict__ h)
{
  long long idx = (long long)blockIdx.x * blockDim.x + threadIdx.x;
  if (idx >= (long long)NNODES * HID) return;
  int c = (int)(idx & 127);
  float mean = bsum[c] * (1.f / NNODES);
  float var  = bsq[c] * (1.f / NNODES) - mean * mean;
  float v = (y[idx] - mean) * rsqrtf(var + BN_EPS) * g[c] + b[c];
  h[idx] = fmaxf(v, 0.f);
}

__global__ void pool_sum(const float* __restrict__ h, const long long* __restrict__ batch,
                         float* __restrict__ gsum, float* __restrict__ cnt)
{
  long long idx = (long long)blockIdx.x * blockDim.x + threadIdx.x;
  if (idx >= (long long)NNODES * HID) return;
  int n = (int)(idx >> 7), c = (int)(idx & 127);
  long long g = batch[n];
  atomicAdd(&gsum[g * HID + c], h[idx]);
  if (c == 0) atomicAdd(&cnt[g], 1.f);
}

__global__ void pool_div(const float* __restrict__ gsum, const float* __restrict__ cnt,
                         float* __restrict__ gpool)
{
  int idx = blockIdx.x * blockDim.x + threadIdx.x;
  if (idx >= NGRAPH * HID) return;
  int g = idx >> 7;
  gpool[idx] = gsum[idx] / fmaxf(cnt[g], 1.f);
}

// naive linear for small / odd shapes (K=3..384, N=1..128)
__global__ void small_linear(const float* __restrict__ A, const float* __restrict__ W,
                             const float* __restrict__ b, float* __restrict__ out,
                             int M, int K, int N, int relu)
{
  long long idx = (long long)blockIdx.x * blockDim.x + threadIdx.x;
  if (idx >= (long long)M * N) return;
  int m = (int)(idx / N), n = (int)(idx % N);
  float s = b[n];
  for (int k = 0; k < K; ++k) s += A[(long)m * K + k] * W[(long)k * N + n];
  if (relu) s = fmaxf(s, 0.f);
  out[idx] = s;
}

__global__ void concat3(const float* __restrict__ a, const float* __restrict__ b,
                        const float* __restrict__ c, float* __restrict__ out)
{
  int idx = blockIdx.x * blockDim.x + threadIdx.x;
  if (idx >= NGRAPH * 384) return;
  int g = idx / 384, j = idx % 384;
  float v = (j < 128) ? a[g * 128 + j] : (j < 256) ? b[g * 128 + j - 128]
                                                   : c[g * 128 + j - 256];
  out[idx] = v;
}

// softmax over 3 attention logits, weighted sum of the three projections
__global__ void attn_fuse(const float* __restrict__ gx, const float* __restrict__ gp,
                          const float* __restrict__ gc, const float* __restrict__ logits,
                          float* __restrict__ fused)
{
  int idx = blockIdx.x * blockDim.x + threadIdx.x;
  if (idx >= NGRAPH * HID) return;
  int g = idx >> 7;
  float a0 = logits[g * 3 + 0], a1 = logits[g * 3 + 1], a2 = logits[g * 3 + 2];
  float m = fmaxf(a0, fmaxf(a1, a2));
  float e0 = __expf(a0 - m), e1 = __expf(a1 - m), e2 = __expf(a2 - m);
  float inv = 1.f / (e0 + e1 + e2);
  fused[idx] = (gx[idx] * e0 + gp[idx] * e1 + gc[idx] * e2) * inv;
}

// ---------------------------------------------------------------------------
static inline unsigned cdiv(long long a, long long b) { return (unsigned)((a + b - 1) / b); }

extern "C" void kernel_launch(void* const* d_in, const int* in_sizes, int n_in,
                              void* d_out, int out_size, void* d_ws, size_t ws_size,
                              hipStream_t stream) {
  (void)in_sizes; (void)n_in; (void)out_size; (void)ws_size;

  // ---- inputs (setup_inputs() insertion order, pytree leaves in order) ----
  const float*     x       = (const float*)d_in[0];        // [100000,64]
  const float*     eattr   = (const float*)d_in[1];        // [600000,3]
  const float*     phys    = (const float*)d_in[2];        // [4096,4]
  const float*     cate    = (const float*)d_in[3];        // [4096,20]
  const long long* eidx    = (const long long*)d_in[4];    // [2,600000] int64
  const long long* batch   = (const long long*)d_in[5];    // [100000]   int64
  const long long* src = eidx, *dst = eidx + NEDGES;

  const float* em_W1 = (const float*)d_in[6];  const float* em_b1 = (const float*)d_in[7];
  const float* em_W2 = (const float*)d_in[8];  const float* em_b2 = (const float*)d_in[9];
  const int linW_i[3] = {10,17,24}, linb_i[3] = {11,18,25}, eps_i[3] = {12,19,26};
  const int mW1_i[3] = {13,20,27},  mb1_i[3] = {14,21,28};
  const int mW2_i[3] = {15,22,29},  mb2_i[3] = {16,23,30};
  const int bng_i[3] = {31,33,35},  bnb_i[3] = {32,34,36};
  const float* phys_W = (const float*)d_in[37]; const float* phys_b = (const float*)d_in[38];
  const float* cat_W  = (const float*)d_in[39]; const float* cat_b  = (const float*)d_in[40];
  const float* px_W = (const float*)d_in[41]; const float* px_b = (const float*)d_in[42];
  const float* pp_W = (const float*)d_in[43]; const float* pp_b = (const float*)d_in[44];
  const float* pc_W = (const float*)d_in[45]; const float* pc_b = (const float*)d_in[46];
  const float* at_W1 = (const float*)d_in[47]; const float* at_b1 = (const float*)d_in[48];
  const float* at_W2 = (const float*)d_in[49]; const float* at_b2 = (const float*)d_in[50];
  const float* cl_W1 = (const float*)d_in[51]; const float* cl_b1 = (const float*)d_in[52];
  const float* cl_W2 = (const float*)d_in[53]; const float* cl_b2 = (const float*)d_in[54];

  // ---- workspace carve ----
  char* ws = (char*)d_ws; size_t off = 0;
  auto alloc = [&](size_t bytes) -> char* {
    char* p = ws + off; off = (off + bytes + 255) & ~(size_t)255; return p;
  };
  unsigned short* ea_bf  = (unsigned short*)alloc((size_t)NEDGES * HID * 2);
  char* scratchE = alloc((size_t)NEDGES * HID * 2);          // h1 first, reused below
  unsigned short* h1_bf  = (unsigned short*)scratchE;
  float*          u_buf  = (float*)scratchE;                               // 51.2MB
  float*          y_buf  = (float*)(scratchE + (size_t)NNODES * HID * 4);  // 51.2MB
  unsigned short* t_bf   = (unsigned short*)(scratchE + (size_t)2 * NNODES * HID * 4);
  float* ea_lin = (float*)alloc((size_t)NEDGES * HID * 4);
  float* hA     = (float*)alloc((size_t)NNODES * HID * 4);
  float* agg    = (float*)alloc((size_t)NNODES * HID * 4);
  float* gsum   = (float*)alloc((size_t)NGRAPH * HID * 4);
  float* cnt    = (float*)alloc((size_t)NGRAPH * 4);
  float* gpool  = (float*)alloc((size_t)NGRAPH * HID * 4);
  float* pbuf   = (float*)alloc((size_t)NGRAPH * 64 * 4);
  float* cbuf   = (float*)alloc((size_t)NGRAPH * 64 * 4);
  float* gx     = (float*)alloc((size_t)NGRAPH * HID * 4);
  float* gp     = (float*)alloc((size_t)NGRAPH * HID * 4);
  float* gc     = (float*)alloc((size_t)NGRAPH * HID * 4);
  float* cat3   = (float*)alloc((size_t)NGRAPH * 384 * 4);
  float* ath    = (float*)alloc((size_t)NGRAPH * HID * 4);
  float* atl    = (float*)alloc((size_t)NGRAPH * 3 * 4);
  float* fusedb = (float*)alloc((size_t)NGRAPH * HID * 4);
  float* clsh   = (float*)alloc((size_t)NGRAPH * HID * 4);
  float* bnsum  = (float*)alloc(512);
  float* bnsq   = (float*)alloc(512);

  // ---- edge MLP: ea = W2 @ relu(W1 @ eattr + b1) + b2, stored bf16 ----
  edge_layer1<<<cdiv((long long)NEDGES * HID, 256), 256, 0, stream>>>(eattr, em_W1, em_b1, h1_bf);
  wmma_gemm_bias<1,1,0><<<cdiv(NEDGES, 64), 128, 0, stream>>>(h1_bf, em_W2, em_b2, ea_bf,
                                                              NEDGES, HID, HID);

  // ---- 3 GINE convs ----
  const float* xin = x;
  for (int i = 0; i < 3; ++i) {
    const int din = (i == 0) ? 64 : 128;
    const float* linW = (const float*)d_in[linW_i[i]];
    const float* linb = (const float*)d_in[linb_i[i]];
    const float* eps  = (const float*)d_in[eps_i[i]];
    // ea_lin = ea @ linW + linb   [E, din]
    wmma_gemm_bias<1,0,0><<<cdiv(NEDGES, 64), 128, 0, stream>>>(ea_bf, linW, linb, ea_lin,
                                                                NEDGES, HID, din);
    hipMemsetAsync(agg, 0, (size_t)NNODES * din * 4, stream);
    msg_scatter<<<cdiv((long long)NEDGES * din, 256), 256, 0, stream>>>(xin, ea_lin, src, dst,
                                                                        agg, din);
    make_u<<<cdiv((long long)NNODES * din, 256), 256, 0, stream>>>(xin, agg, eps, u_buf,
                                                                   (long long)NNODES * din);
    // node MLP: relu(u@W1+b1) @ W2 + b2
    wmma_gemm_bias<0,1,1><<<cdiv(NNODES, 64), 128, 0, stream>>>(
        u_buf, (const float*)d_in[mW1_i[i]], (const float*)d_in[mb1_i[i]], t_bf,
        NNODES, din, HID);
    wmma_gemm_bias<1,0,0><<<cdiv(NNODES, 64), 128, 0, stream>>>(
        t_bf, (const float*)d_in[mW2_i[i]], (const float*)d_in[mb2_i[i]], y_buf,
        NNODES, HID, HID);
    // batchnorm (training stats) + relu
    hipMemsetAsync(bnsum, 0, 512, stream);
    hipMemsetAsync(bnsq,  0, 512, stream);
    bn_stats<<<256, 256, 0, stream>>>(y_buf, bnsum, bnsq);
    bn_apply<<<cdiv((long long)NNODES * HID, 256), 256, 0, stream>>>(
        y_buf, bnsum, bnsq, (const float*)d_in[bng_i[i]], (const float*)d_in[bnb_i[i]], hA);
    xin = hA;
  }

  // ---- global mean pool ----
  hipMemsetAsync(gsum, 0, (size_t)NGRAPH * HID * 4, stream);
  hipMemsetAsync(cnt,  0, (size_t)NGRAPH * 4, stream);
  pool_sum<<<cdiv((long long)NNODES * HID, 256), 256, 0, stream>>>(hA, batch, gsum, cnt);
  pool_div<<<cdiv(NGRAPH * HID, 256), 256, 0, stream>>>(gsum, cnt, gpool);

  // ---- graph-level heads ----
  small_linear<<<cdiv((long long)NGRAPH * 64, 256), 256, 0, stream>>>(phys, phys_W, phys_b,
                                                                      pbuf, NGRAPH, 4, 64, 1);
  small_linear<<<cdiv((long long)NGRAPH * 64, 256), 256, 0, stream>>>(cate, cat_W, cat_b,
                                                                      cbuf, NGRAPH, 20, 64, 1);
  wmma_gemm_bias<0,0,0><<<cdiv(NGRAPH, 64), 128, 0, stream>>>(gpool, px_W, px_b, gx,
                                                              NGRAPH, 128, 128);
  wmma_gemm_bias<0,0,0><<<cdiv(NGRAPH, 64), 128, 0, stream>>>(pbuf, pp_W, pp_b, gp,
                                                              NGRAPH, 64, 128);
  wmma_gemm_bias<0,0,0><<<cdiv(NGRAPH, 64), 128, 0, stream>>>(cbuf, pc_W, pc_b, gc,
                                                              NGRAPH, 64, 128);
  concat3<<<cdiv(NGRAPH * 384, 256), 256, 0, stream>>>(gx, gp, gc, cat3);
  small_linear<<<cdiv((long long)NGRAPH * 128, 256), 256, 0, stream>>>(cat3, at_W1, at_b1,
                                                                       ath, NGRAPH, 384, 128, 1);
  small_linear<<<cdiv((long long)NGRAPH * 3, 256), 256, 0, stream>>>(ath, at_W2, at_b2,
                                                                     atl, NGRAPH, 128, 3, 0);
  attn_fuse<<<cdiv(NGRAPH * HID, 256), 256, 0, stream>>>(gx, gp, gc, atl, fusedb);
  wmma_gemm_bias<0,0,1><<<cdiv(NGRAPH, 64), 128, 0, stream>>>(fusedb, cl_W1, cl_b1, clsh,
                                                              NGRAPH, 128, 128);
  small_linear<<<cdiv((long long)NGRAPH, 256), 256, 0, stream>>>(clsh, cl_W2, cl_b2,
                                                                 (float*)d_out, NGRAPH, 128, 1, 0);
}
